// LoRA_MOE_QK_28381143892014
// MI455X (gfx1250) — matrix operand order repaired and checked by reading, not compiled, not run on hardware
//
#include <hip/hip_runtime.h>
#include <hip/hip_bf16.h>
#include <math.h>

// ---------- problem constants ----------
#define BATCH   4
#define SEQ     2048
#define DIM     4096          // D (K of base gemm)
#define OUT     4096          // O (N of base gemm)
#define NEXP    8
#define RANK    16
#define ER      128           // NEXP*RANK
#define MTOT    (BATCH*SEQ)   // 8192
#define QS      611           // 34+576+1
#define QCOUNT  1436          // 2047-611
#define SCALING 2.0f          // 32/16

typedef __bf16 bf16;
typedef __attribute__((ext_vector_type(16))) __bf16 v16bf;
typedef __attribute__((ext_vector_type(8)))  float  v8f;

union FragBF { uint4 q[2]; v16bf v; };

#define LDS_STRIDE 40   // 32 bf16 payload + 8 pad; 80B row stays 16B aligned

// =====================================================================
// Router: q = mean(x[:, 611:2047, :]) ; logits = q@rW.T + rb ; softmax
// one block per batch
// =====================================================================
__global__ void router_kernel(const float* __restrict__ x,
                              const float* __restrict__ rW,
                              const float* __restrict__ rb,
                              float* __restrict__ routing) {
    const int b = blockIdx.x;
    const int tid = threadIdx.x;
    float q[16];
#pragma unroll
    for (int i = 0; i < 16; ++i) q[i] = 0.0f;

    const float* xb = x + (size_t)b * SEQ * DIM;
    for (int s = QS; s < SEQ - 1; ++s) {
        const float* xp = xb + (size_t)s * DIM;
#pragma unroll
        for (int i = 0; i < 16; ++i) q[i] += xp[tid + 256 * i];
    }
    const float inv = 1.0f / (float)QCOUNT;

    __shared__ float red[256];
    __shared__ float logits[NEXP];
    for (int e = 0; e < NEXP; ++e) {
        const float* rwp = rW + (size_t)e * DIM;
        float a = 0.0f;
#pragma unroll
        for (int i = 0; i < 16; ++i) a += q[i] * rwp[tid + 256 * i];
        red[tid] = a;
        __syncthreads();
        for (int off = 128; off > 0; off >>= 1) {
            if (tid < off) red[tid] += red[tid + off];
            __syncthreads();
        }
        if (tid == 0) logits[e] = red[0] * inv + rb[e];
        __syncthreads();
    }
    if (tid == 0) {
        float mx = logits[0];
        for (int e = 1; e < NEXP; ++e) mx = fmaxf(mx, logits[e]);
        float sum = 0.0f, ex[NEXP];
        for (int e = 0; e < NEXP; ++e) { ex[e] = expf(logits[e] - mx); sum += ex[e]; }
        float isum = 1.0f / sum;
        for (int e = 0; e < NEXP; ++e) routing[b * NEXP + e] = ex[e] * isum;
    }
}

// =====================================================================
// elementwise f32 -> bf16
// =====================================================================
__global__ void cvt_bf16_kernel(const float* __restrict__ src,
                                bf16* __restrict__ dst, int n) {
    int i = blockIdx.x * blockDim.x + threadIdx.x;
    int stride = gridDim.x * blockDim.x;
    for (; i < n; i += stride) dst[i] = (bf16)src[i];
}

// B (E,O,R) -> B_cat (O, ER) bf16
__global__ void bcat_kernel(const float* __restrict__ Bsrc,
                            bf16* __restrict__ dst) {
    int idx = blockIdx.x * blockDim.x + threadIdx.x; // o*128 + er
    int o  = idx >> 7;
    int er = idx & 127;
    int e = er >> 4, r = er & 15;
    dst[idx] = (bf16)Bsrc[(size_t)e * OUT * RANK + (size_t)o * RANK + r];
}

// =====================================================================
// CDNA5 async-copy helpers (GLOBAL_LOAD_ASYNC_TO_LDS_*, ASYNCcnt)
// =====================================================================
__device__ __forceinline__ unsigned lds_off_u32(const void* p) {
    // flat LDS addresses carry the LDS byte offset in addr[31:0]
    return (unsigned)(unsigned long long)p;
}

__device__ __forceinline__ void async_b128(unsigned lds_addr, unsigned voff_bytes,
                                           const bf16* sbase) {
    // GVS mode: mem = SGPR[base] + VGPR[voff]; LDS[vdst] <- 16 bytes
    asm volatile("global_load_async_to_lds_b128 %0, %1, %2"
                 :
                 : "v"(lds_addr), "v"(voff_bytes), "s"(sbase)
                 : "memory");
}

__device__ __forceinline__ void wait_async0() {
    asm volatile("s_wait_asynccnt 0x0" ::: "memory");
}

// stage one 128x32 bf16 tile pair (A rows, B rows) into LDS buffers
__device__ __forceinline__ void async_stage(unsigned sAb, unsigned sBb,
                                            const bf16* Ag, int lda,
                                            const bf16* Bg, int ldb,
                                            int k0, int tid) {
#pragma unroll
    for (int l = 0; l < 2; ++l) {
        int c = tid + 256 * l;       // 0..511 : 128 rows x 4 chunks of 8 bf16
        int r = c >> 2;
        int off = (c & 3) * 8;
        unsigned lofs = (unsigned)(r * LDS_STRIDE + off) * 2u;
        async_b128(sAb + lofs, (unsigned)(r * lda + k0 + off) * 2u, Ag);
        async_b128(sBb + lofs, (unsigned)(r * ldb + k0 + off) * 2u, Bg);
    }
}

__device__ __forceinline__ void load_fragA(FragBF& f, const bf16* sA, int row, int half) {
    const bf16* p = sA + row * LDS_STRIDE;
    f.q[0] = *(const uint4*)&p[half * 8];        // K = half*8 .. +7
    f.q[1] = *(const uint4*)&p[16 + half * 8];   // K = 16+half*8 .. +7
}

__device__ __forceinline__ void load_fragB(FragBF& f, const bf16* sB, int row, int half) {
    const bf16* p = sB + row * LDS_STRIDE;
    f.q[0] = *(const uint4*)&p[half * 8];        // K = half*8 .. +7
    f.q[1] = *(const uint4*)&p[16 + half * 8];   // K = 16+half*8 .. +7
}

__device__ __forceinline__ void mma_tiles(const bf16* sA, const bf16* sB,
                                          v8f acc[4][2],
                                          int wm, int wn, int rrow, int half) {
    FragBF af[4], bfr[2];
#pragma unroll
    for (int mt = 0; mt < 4; ++mt) load_fragA(af[mt], sA, wm * 64 + mt * 16 + rrow, half);
#pragma unroll
    for (int nt = 0; nt < 2; ++nt) load_fragB(bfr[nt], sB, wn * 32 + nt * 16 + rrow, half);
#pragma unroll
    for (int mt = 0; mt < 4; ++mt)
#pragma unroll
        for (int nt = 0; nt < 2; ++nt)
            acc[mt][nt] = __builtin_amdgcn_wmma_f32_16x16x32_bf16(
                false, af[mt].v, false, bfr[nt].v, (short)0, acc[mt][nt], false, false);
}

// double-buffered K loop over one operand pair
__device__ __forceinline__ void gemm_loop(bf16 (*sA)[128 * LDS_STRIDE],
                                          bf16 (*sB)[128 * LDS_STRIDE],
                                          const bf16* Ag, int lda,
                                          const bf16* Bg, int ldb,
                                          int kdim, v8f acc[4][2],
                                          int tid, int wm, int wn, int rrow, int half) {
    const unsigned sA0 = lds_off_u32(&sA[0][0]), sA1 = lds_off_u32(&sA[1][0]);
    const unsigned sB0 = lds_off_u32(&sB[0][0]), sB1 = lds_off_u32(&sB[1][0]);
    const int nk = kdim / 32;

    async_stage(sA0, sB0, Ag, lda, Bg, ldb, 0, tid);
    wait_async0();
    __syncthreads();
#pragma unroll 1
    for (int k = 0; k < nk; ++k) {
        const int cur = k & 1;
        if (k + 1 < nk)
            async_stage(cur ? sA0 : sA1, cur ? sB0 : sB1,
                        Ag, lda, Bg, ldb, (k + 1) * 32, tid);
        mma_tiles(sA[cur], sB[cur], acc, wm, wn, rrow, half);
        wait_async0();
        __syncthreads();
    }
}

// =====================================================================
// u-GEMM: uw[m, er] = (x @ A_cat.T)[m,er] * routing[b(m), er/16] * SCALING
// grid: (1, MTOT/128), block 256
// =====================================================================
__global__ void lora_u_kernel(const bf16* __restrict__ X,
                              const bf16* __restrict__ Ah,   // (ER, DIM)
                              const float* __restrict__ routing,
                              bf16* __restrict__ U) {        // (MTOT, ER)
    __shared__ __align__(16) bf16 sA[2][128 * LDS_STRIDE];
    __shared__ __align__(16) bf16 sB[2][128 * LDS_STRIDE];

    const int tid  = threadIdx.x;
    const int lane = tid & 31;
    const int wv   = tid >> 5;
    const int wm   = wv >> 2;      // 0..1
    const int wn   = wv & 3;       // 0..3
    const int half = lane >> 4;
    const int rrow = lane & 15;
    const int m0   = blockIdx.y * 128;

    v8f acc[4][2];
#pragma unroll
    for (int mt = 0; mt < 4; ++mt)
#pragma unroll
        for (int nt = 0; nt < 2; ++nt) acc[mt][nt] = {};

    gemm_loop(sA, sB, X + (size_t)m0 * DIM, DIM, Ah, DIM, DIM,
              acc, tid, wm, wn, rrow, half);

    // epilogue: scale by routing and store bf16
#pragma unroll
    for (int mt = 0; mt < 4; ++mt) {
#pragma unroll
        for (int nt = 0; nt < 2; ++nt) {
            int er = wn * 32 + nt * 16 + rrow;        // 0..127
            int e  = er >> 4;
#pragma unroll
            for (int i = 0; i < 8; ++i) {
                int m = m0 + wm * 64 + mt * 16 + half * 8 + i;
                int bb = m >> 11;                      // m / SEQ
                float v = acc[mt][nt][i] * routing[bb * NEXP + e] * SCALING;
                U[(size_t)m * ER + er] = (bf16)v;
            }
        }
    }
}

// =====================================================================
// main GEMM: out = x@W.T + bias + uw@B_cat.T
// grid: (OUT/128, MTOT/128), block 256
// =====================================================================
__global__ void main_gemm_kernel(const bf16* __restrict__ X,     // (MTOT, DIM)
                                 const bf16* __restrict__ Wh,    // (OUT, DIM)
                                 const float* __restrict__ bias, // (OUT)
                                 const bf16* __restrict__ U,     // (MTOT, ER)
                                 const bf16* __restrict__ Bh,    // (OUT, ER)
                                 float* __restrict__ out) {
    __shared__ __align__(16) bf16 sA[2][128 * LDS_STRIDE];
    __shared__ __align__(16) bf16 sB[2][128 * LDS_STRIDE];

    const int tid  = threadIdx.x;
    const int lane = tid & 31;
    const int wv   = tid >> 5;
    const int wm   = wv >> 2;
    const int wn   = wv & 3;
    const int half = lane >> 4;
    const int rrow = lane & 15;
    const int n0   = blockIdx.x * 128;
    const int m0   = blockIdx.y * 128;

    v8f acc[4][2];
#pragma unroll
    for (int mt = 0; mt < 4; ++mt)
#pragma unroll
        for (int nt = 0; nt < 2; ++nt) acc[mt][nt] = {};

    // phase 1: K = DIM over x / W
    gemm_loop(sA, sB, X + (size_t)m0 * DIM, DIM, Wh + (size_t)n0 * DIM, DIM, DIM,
              acc, tid, wm, wn, rrow, half);
    // phase 2: K = ER over uw / B_cat (rank-128 LoRA fused)
    gemm_loop(sA, sB, U + (size_t)m0 * ER, ER, Bh + (size_t)n0 * ER, ER, ER,
              acc, tid, wm, wn, rrow, half);

    // epilogue: add bias, store f32
#pragma unroll
    for (int nt = 0; nt < 2; ++nt) {
        int n = n0 + wn * 32 + nt * 16 + rrow;
        float bn = bias[n];
#pragma unroll
        for (int mt = 0; mt < 4; ++mt) {
#pragma unroll
            for (int i = 0; i < 8; ++i) {
                int m = m0 + wm * 64 + mt * 16 + half * 8 + i;
                out[(size_t)m * OUT + n] = acc[mt][nt][i] + bn;
            }
        }
    }
}

// =====================================================================
// launcher
// =====================================================================
extern "C" void kernel_launch(void* const* d_in, const int* in_sizes, int n_in,
                              void* d_out, int out_size, void* d_ws, size_t ws_size,
                              hipStream_t stream) {
    const float* x   = (const float*)d_in[0];  // (4,2048,4096)
    const float* W   = (const float*)d_in[1];  // (4096,4096)
    const float* b   = (const float*)d_in[2];  // (4096)
    const float* A   = (const float*)d_in[3];  // (8,16,4096) == flat (128,4096)
    const float* Bm  = (const float*)d_in[4];  // (8,4096,16)
    const float* rW  = (const float*)d_in[5];  // (8,4096)
    const float* rb  = (const float*)d_in[6];  // (8)
    float* out = (float*)d_out;

    // workspace layout (256B aligned chunks)
    char* ws = (char*)d_ws;
    float* routing = (float*)ws;                                   // 32 f
    bf16*  xh = (bf16*)(ws + 4096);                                // 64 MB
    bf16*  wh = (bf16*)((char*)xh + (size_t)MTOT * DIM * 2);       // 32 MB
    bf16*  ah = (bf16*)((char*)wh + (size_t)OUT  * DIM * 2);       // 1 MB
    bf16*  bh = (bf16*)((char*)ah + (size_t)ER   * DIM * 2);       // 1 MB
    bf16*  uw = (bf16*)((char*)bh + (size_t)OUT  * ER  * 2);       // 2 MB

    // 1) router
    router_kernel<<<BATCH, 256, 0, stream>>>(x, rW, rb, routing);

    // 2) precision conversions
    cvt_bf16_kernel<<<8192, 256, 0, stream>>>(x, xh, MTOT * DIM);
    cvt_bf16_kernel<<<8192, 256, 0, stream>>>(W, wh, OUT * DIM);
    cvt_bf16_kernel<<<2048, 256, 0, stream>>>(A, ah, ER * DIM);
    bcat_kernel<<<(OUT * ER) / 256, 256, 0, stream>>>(Bm, bh);

    // 3) u = (x @ A_cat.T) * routing * scaling   -> bf16
    lora_u_kernel<<<dim3(1, MTOT / 128), 256, 0, stream>>>(xh, ah, routing, uw);

    // 4) out = x@W.T + bias + u@B_cat.T
    main_gemm_kernel<<<dim3(OUT / 128, MTOT / 128), 256, 0, stream>>>(xh, wh, b, uw, bh, out);
}
